// ViTMoE_55490977464340
// MI455X (gfx1250) — matrix operand order repaired and checked by reading, not compiled, not run on hardware
//
#include <hip/hip_runtime.h>

// ---------------------------------------------------------------------------
// ViT-MoE forward for MI455X (gfx1250, wave32, WMMA).
// Precision: bf16 WMMA (v_wmma_f32_16x16x32_bf16) with fp32 accumulate.
// MoE: deterministic token routing -> grouped per-expert GEMMs.
// GEMM: 128x128 tile, 8 waves, register-staged pipeline with ping-pong LDS
//       (single barrier per K-step), float4 global loads, packed bf16 LDS
//       stores, global_prefetch for K+2 tiles.
// ---------------------------------------------------------------------------

typedef __attribute__((ext_vector_type(16))) __bf16 v16bf;
typedef __attribute__((ext_vector_type(8)))  float  v8f;

#define DEVFN static __device__ __forceinline__

static constexpr int kD     = 768;
static constexpr int kNH    = 12;
static constexpr int kHD    = 64;
static constexpr int kSEQ   = 197;
static constexpr int kB     = 8;
static constexpr int kT     = kB * kSEQ;     // 1576 tokens
static constexpr int kHID   = 3072;
static constexpr int kE     = 8;
static constexpr int kNCLS  = 1000;
static constexpr int kNP    = 196;

// cheap fp32->bf16 (round-half-up): 1 add + 1 shift
DEVFN unsigned bf16_bits(float f) {
  return (__float_as_uint(f) + 0x8000u) >> 16;
}
// pack two fp32 -> two bf16 in one u32
DEVFN unsigned pack2_bf16(float lo, float hi) {
  unsigned a = (__float_as_uint(lo) + 0x8000u) >> 16;
  unsigned b = (__float_as_uint(hi) + 0x8000u) & 0xFFFF0000u;
  return a | b;
}
DEVFN __bf16 f2bf(float f) {
  unsigned short us = (unsigned short)bf16_bits(f);
  __bf16 b;
  __builtin_memcpy(&b, &us, 2);
  return b;
}

DEVFN float gelu_exact(float x) {
  return 0.5f * x * (1.0f + erff(x * 0.7071067811865476f));
}

// ---------------------------------------------------------------------------
// Generic tiled WMMA GEMM:  C = act(alpha * A@B + bias) [+ residual]
// MODE: 0 = plain rows, 1 = gather A rows via row_idx (MoE gemm1 / head),
//       2 = rows offset by moff[z]   (MoE gemm2).
// Batched via blockIdx.z with offsets decomposed as z = bb*div + ss.
// ---------------------------------------------------------------------------
struct GemmParams {
  const float* A;
  const float* Bm;
  const float* bias;      // may be null
  const float* R;         // residual, may be null
  float*       C;
  const int*   row_idx;   // mode 1
  const int*   mcnt;      // per-z M override (device), may be null
  const int*   moff;      // per-z row base (device), may be null
  long long lda, ldb, ldc, ldr;
  long long ab_b, ab_s, bb_b, bb_s, cb_b, cb_s, rb_b, rb_s, bias_b;
  int M, N, K;
  int div;                // z decomposition divisor
  int transB;             // dispatch only
  int act;                // dispatch only
  int mode;               // dispatch only
  float alpha;
};

template <int MODE, int TRANSB, int ACT>
__global__ __launch_bounds__(256) void gemm_wmma_kernel(GemmParams p) {
  // ping-pong LDS: one barrier per K-step
  __shared__ __bf16 As[2][128][40];   // padded K stride -> conflict-free b128 reads
  __shared__ __bf16 Bs[2][128][40];   // stored [n][k]

  const int z  = blockIdx.z;
  const int bb = z / p.div;
  const int ss = z % p.div;
  const float* __restrict__ A  = p.A  + (long long)bb * p.ab_b + (long long)ss * p.ab_s;
  const float* __restrict__ Bg = p.Bm + (long long)bb * p.bb_b + (long long)ss * p.bb_s;
  float* __restrict__ C        = p.C  + (long long)bb * p.cb_b + (long long)ss * p.cb_s;

  const int M    = p.mcnt ? p.mcnt[z] : p.M;
  const int base = p.moff ? p.moff[z] : 0;
  const int tileM = blockIdx.y * 128;
  const int tileN = blockIdx.x * 128;
  if (tileM >= M) return;   // uniform per block

  const int tid  = threadIdx.x;
  const int lane = tid & 31;
  const int wave = tid >> 5;
  const int wm   = wave >> 1;      // 0..3  (rows, 32 each)
  const int wn   = wave & 1;       // 0..1  (cols, 64 each)
  const int half = lane >> 4;
  const int lm   = lane & 15;

  const bool aVec = ((p.lda & 3) == 0);   // float4-able A stride
  const bool bVec = ((p.ldb & 3) == 0);   // float4-able B stride

  // ---- per-thread staging slots (4 float4 for A, 4 for B) ----
  int rA[4], kqA[4];
  int rB[4], qB[4];
#pragma unroll
  for (int i = 0; i < 4; ++i) {
    int idx4 = i * 256 + tid;
    rA[i] = idx4 >> 3;
    kqA[i] = (idx4 & 7) * 4;
    if (TRANSB) { rB[i] = idx4 >> 3; qB[i] = (idx4 & 7) * 4; }
    else        { rB[i] = idx4 >> 5; qB[i] = (idx4 & 31) * 4; }
  }
  // hoist gathered A row offsets (K-invariant)
  long long arow[4];
#pragma unroll
  for (int i = 0; i < 4; ++i) {
    int gr = tileM + rA[i];
    long long r = 0;
    if (gr < M) {
      if (MODE == 1)      r = p.row_idx[base + gr];
      else if (MODE == 2) r = (long long)base + gr;
      else                r = gr;
    }
    arow[i] = r * p.lda;
  }

  auto loadA = [&](int kt, float4* aR) {
    const int k0 = kt * 32;
    const bool fullK = (k0 + 32 <= p.K);
#pragma unroll
    for (int i = 0; i < 4; ++i) {
      float4 v = {0.0f, 0.0f, 0.0f, 0.0f};
      int gr = tileM + rA[i];
      int gk = k0 + kqA[i];
      if (gr < M) {
        if (aVec && fullK) {
          v = *(const float4*)(A + arow[i] + gk);
        } else {
#pragma unroll
          for (int j = 0; j < 4; ++j)
            if (gk + j < p.K) ((float*)&v)[j] = A[arow[i] + gk + j];
        }
      }
      aR[i] = v;
    }
  };
  auto storeA = [&](int buf, const float4* aR) {
#pragma unroll
    for (int i = 0; i < 4; ++i) {
      uint2 w;
      w.x = pack2_bf16(aR[i].x, aR[i].y);
      w.y = pack2_bf16(aR[i].z, aR[i].w);
      *(uint2*)(&As[buf][rA[i]][kqA[i]]) = w;   // 8B aligned (kqA mult of 4, row 80B)
    }
  };

  auto loadB = [&](int kt, float4* bR) {
    const int k0 = kt * 32;
    const bool fullK = (k0 + 32 <= p.K);
#pragma unroll
    for (int i = 0; i < 4; ++i) {
      float4 v = {0.0f, 0.0f, 0.0f, 0.0f};
      if (TRANSB) {
        int gn = tileN + rB[i];
        int gk = k0 + qB[i];
        if (gn < p.N) {
          if (bVec && fullK) {
            v = *(const float4*)(Bg + (long long)gn * p.ldb + gk);
          } else {
#pragma unroll
            for (int j = 0; j < 4; ++j)
              if (gk + j < p.K) ((float*)&v)[j] = Bg[(long long)gn * p.ldb + gk + j];
          }
        }
      } else {
        int gk = k0 + rB[i];
        int gn = tileN + qB[i];
        if (gk < p.K) {
          if (bVec && (gn + 3 < p.N)) {
            v = *(const float4*)(Bg + (long long)gk * p.ldb + gn);
          } else {
#pragma unroll
            for (int j = 0; j < 4; ++j)
              if (gn + j < p.N) ((float*)&v)[j] = Bg[(long long)gk * p.ldb + gn + j];
          }
        }
      }
      bR[i] = v;
    }
  };
  auto storeB = [&](int buf, const float4* bR) {
#pragma unroll
    for (int i = 0; i < 4; ++i) {
      if (TRANSB) {
        uint2 w;
        w.x = pack2_bf16(bR[i].x, bR[i].y);
        w.y = pack2_bf16(bR[i].z, bR[i].w);
        *(uint2*)(&Bs[buf][rB[i]][qB[i]]) = w;
      } else {
#pragma unroll
        for (int j = 0; j < 4; ++j)
          Bs[buf][qB[i] + j][rB[i]] = f2bf(((const float*)&bR[i])[j]);
      }
    }
  };

  v8f acc[2][4];
#pragma unroll
  for (int mi = 0; mi < 2; ++mi)
#pragma unroll
    for (int ni = 0; ni < 4; ++ni)
#pragma unroll
      for (int r = 0; r < 8; ++r) acc[mi][ni][r] = 0.0f;

  const int nk = (p.K + 31) >> 5;
  float4 aR[4], bR[4];
  loadA(0, aR); loadB(0, bR);
  storeA(0, aR); storeB(0, bR);
  __syncthreads();

  int cur = 0;
  for (int kt = 0; kt < nk; ++kt) {
    const bool more = (kt + 1 < nk);
    if (more) { loadA(kt + 1, aR); loadB(kt + 1, bR); }  // overlap with WMMA

    // keep the memory system two tiles ahead (global_prefetch_b8)
    if (kt + 2 < nk) {
      const int k2 = (kt + 2) * 32;
#pragma unroll
      for (int i = 0; i < 4; ++i)
        if (tileM + rA[i] < M)
          __builtin_prefetch(A + arow[i] + k2 + kqA[i], 0, 1);
#pragma unroll
      for (int i = 0; i < 4; ++i) {
        if (TRANSB) {
          if (tileN + rB[i] < p.N)
            __builtin_prefetch(Bg + (long long)(tileN + rB[i]) * p.ldb + k2 + qB[i], 0, 1);
        } else {
          if (k2 + rB[i] < p.K && tileN + qB[i] < p.N)
            __builtin_prefetch(Bg + (long long)(k2 + rB[i]) * p.ldb + tileN + qB[i], 0, 1);
        }
      }
    }

    // ---- fragments, A and B both in contiguous k order (k = 16*half + e) ----
    v16bf af[2], bfv[4];
#pragma unroll
    for (int mi = 0; mi < 2; ++mi) {
      int m = wm * 32 + mi * 16 + lm;
#pragma unroll
      for (int v = 0; v < 8; ++v) {
        // 16-bit A 16x32: VGPR v -> K = 2v + 16*half (matches B contiguous order)
        int kb = 2 * v + 16 * half;
        af[mi][2 * v]     = As[cur][m][kb];
        af[mi][2 * v + 1] = As[cur][m][kb + 1];
      }
    }
    const int koff = half * 16;  // B: lanes 0-15 hold K=0..15, lanes 16-31 hold K=16..31
#pragma unroll
    for (int ni = 0; ni < 4; ++ni) {
      int n = wn * 64 + ni * 16 + lm;
#pragma unroll
      for (int e = 0; e < 16; ++e) bfv[ni][e] = Bs[cur][n][koff + e];
    }

#pragma unroll
    for (int mi = 0; mi < 2; ++mi)
#pragma unroll
      for (int ni = 0; ni < 4; ++ni)
        acc[mi][ni] = __builtin_amdgcn_wmma_f32_16x16x32_bf16(
            false, af[mi], false, bfv[ni], (short)0, acc[mi][ni], false, false);

    if (more) {
      storeA(cur ^ 1, aR); storeB(cur ^ 1, bR);   // other buffer: no reader conflict
      __syncthreads();                            // single barrier per K-step
      cur ^= 1;
    }
  }

  // ---- epilogue: C layout VGPR r -> M = r + 8*half, N = lane&15 ----
  const float* bias = p.bias ? (p.bias + (long long)z * p.bias_b) : nullptr;
  const float* R    = p.R ? (p.R + (long long)bb * p.rb_b + (long long)ss * p.rb_s) : nullptr;
#pragma unroll
  for (int mi = 0; mi < 2; ++mi)
#pragma unroll
    for (int ni = 0; ni < 4; ++ni)
#pragma unroll
      for (int r = 0; r < 8; ++r) {
        int gm = tileM + wm * 32 + mi * 16 + r + 8 * half;
        int gn = tileN + wn * 64 + ni * 16 + lm;
        if (gm < M && gn < p.N) {
          float v = acc[mi][ni][r] * p.alpha;
          if (bias) v += bias[gn];
          if (ACT == 1) v = gelu_exact(v);
          long long crow = (MODE != 0) ? (long long)(base + gm) : (long long)gm;
          if (R) v += R[crow * p.ldr + gn];
          C[crow * p.ldc + gn] = v;
        }
      }
}

// ---------------------------------------------------------------------------
// Support kernels
// ---------------------------------------------------------------------------
__global__ void patchify_kernel(const float* __restrict__ x, float* __restrict__ out) {
  int b = blockIdx.y;
  int pidx = blockIdx.x;                 // 0..195
  int gy = pidx / 14, gx = pidx % 14;
  for (int c = threadIdx.x; c < kD; c += 256) {
    int ch = c >> 8;                     // channel (256 = P*P)
    int r  = c & 255;
    int py = r >> 4, px = r & 15;
    float v = x[(((long long)b * 3 + ch) * 224 + (gy * 16 + py)) * 224 + (gx * 16 + px)];
    out[((long long)b * kNP + pidx) * kD + c] = v;
  }
}

__global__ void assemble_h_kernel(const float* __restrict__ tok,
                                  const float* __restrict__ cls,
                                  const float* __restrict__ pos,
                                  float* __restrict__ h) {
  int b = blockIdx.y, n = blockIdx.x;    // n in 0..196
  for (int c = threadIdx.x; c < kD; c += 256) {
    float v = (n == 0) ? cls[c] : tok[((long long)b * kNP + (n - 1)) * kD + c];
    h[((long long)b * kSEQ + n) * kD + c] = v + pos[(long long)n * kD + c];
  }
}

__global__ __launch_bounds__(256) void layernorm_kernel(const float* __restrict__ X,
                                                        const float* __restrict__ g,
                                                        const float* __restrict__ bt,
                                                        float* __restrict__ Y) {
  __shared__ float red[256];
  long long row = blockIdx.x;
  const float* x = X + row * kD;
  float* y = Y + row * kD;
  int t = threadIdx.x;
  float v0 = x[t], v1 = x[t + 256], v2 = x[t + 512];
  red[t] = v0 + v1 + v2;
  __syncthreads();
  for (int s = 128; s > 0; s >>= 1) { if (t < s) red[t] += red[t + s]; __syncthreads(); }
  float m = red[0] * (1.0f / (float)kD);
  __syncthreads();
  float d0 = v0 - m, d1 = v1 - m, d2 = v2 - m;
  red[t] = d0 * d0 + d1 * d1 + d2 * d2;
  __syncthreads();
  for (int s = 128; s > 0; s >>= 1) { if (t < s) red[t] += red[t + s]; __syncthreads(); }
  float rs = rsqrtf(red[0] * (1.0f / (float)kD) + 1e-5f);
  y[t]       = d0 * rs * g[t]       + bt[t];
  y[t + 256] = d1 * rs * g[t + 256] + bt[t + 256];
  y[t + 512] = d2 * rs * g[t + 512] + bt[t + 512];
}

__global__ __launch_bounds__(256) void softmax_kernel(float* __restrict__ S, int n) {
  __shared__ float red[256];
  long long row = (long long)blockIdx.y * gridDim.x + blockIdx.x;
  float* x = S + row * n;
  int t = threadIdx.x;
  float v = (t < n) ? x[t] : -3.4e38f;
  red[t] = v;
  __syncthreads();
  for (int s = 128; s > 0; s >>= 1) { if (t < s) red[t] = fmaxf(red[t], red[t + s]); __syncthreads(); }
  float mx = red[0];
  __syncthreads();
  float e = (t < n) ? expf(v - mx) : 0.0f;
  red[t] = e;
  __syncthreads();
  for (int s = 128; s > 0; s >>= 1) { if (t < s) red[t] += red[t + s]; __syncthreads(); }
  float inv = 1.0f / red[0];
  if (t < n) x[t] = e * inv;
}

// wave-per-token: 8 gate logits, top-2, softmax over the two selected logits
__global__ __launch_bounds__(32) void gate_topk_kernel(const float* __restrict__ X,
                                                       const float* __restrict__ wg,
                                                       int* __restrict__ topi,
                                                       float* __restrict__ topg) {
  int t = blockIdx.x;
  int lane = threadIdx.x;
  const float* x = X + (long long)t * kD;
  float acc[kE];
#pragma unroll
  for (int e = 0; e < kE; ++e) acc[e] = 0.0f;
  for (int j = lane; j < kD; j += 32) {
    float xv = x[j];
#pragma unroll
    for (int e = 0; e < kE; ++e) acc[e] += xv * wg[j * kE + e];
  }
#pragma unroll
  for (int e = 0; e < kE; ++e)
    for (int off = 16; off > 0; off >>= 1) acc[e] += __shfl_down(acc[e], off);
  if (lane == 0) {
    int i0 = 0; float v0 = acc[0];
    for (int e = 1; e < kE; ++e) if (acc[e] > v0) { v0 = acc[e]; i0 = e; }
    int i1 = -1; float v1 = -3.4e38f;
    for (int e = 0; e < kE; ++e) if (e != i0 && acc[e] > v1) { v1 = acc[e]; i1 = e; }
    float e1 = expf(v1 - v0);         // v0 is max
    float s = 1.0f + e1;
    topi[2 * t] = i0;       topi[2 * t + 1] = i1;
    topg[2 * t] = 1.0f / s; topg[2 * t + 1] = e1 / s;
  }
}

// Deterministic routing: wave e builds the compacted token list of expert e
// using ballot + prefix popcount. One workgroup, 8 waves.
__global__ __launch_bounds__(256) void route_kernel(const int* __restrict__ topi,
                                                    int* __restrict__ rowlist,
                                                    int* __restrict__ rowpos,
                                                    int* __restrict__ mcnt,
                                                    int* __restrict__ moff) {
  __shared__ int cnts[kE];
  __shared__ int soff[kE];
  int lane = threadIdx.x & 31;
  int e = threadIdx.x >> 5;            // expert = wave id, 0..7
  // pass 1: count
  int cnt = 0;
  for (int base = 0; base < kT; base += 32) {
    int t = base + lane;
    bool m = false;
    if (t < kT) m = (topi[2 * t] == e) || (topi[2 * t + 1] == e);
    unsigned mask = (unsigned)__ballot(m);
    cnt += __popc(mask);
  }
  if (lane == 0) cnts[e] = cnt;
  __syncthreads();
  if (threadIdx.x == 0) {
    int off = 0;
    for (int i = 0; i < kE; ++i) { soff[i] = off; moff[i] = off; mcnt[i] = cnts[i]; off += cnts[i]; }
  }
  __syncthreads();
  // pass 2: fill
  int pos = soff[e];
  for (int base = 0; base < kT; base += 32) {
    int t = base + lane;
    bool m = false;
    int s = 0;
    if (t < kT) {
      if (topi[2 * t] == e) { m = true; s = 0; }
      else if (topi[2 * t + 1] == e) { m = true; s = 1; }
    }
    unsigned mask = (unsigned)__ballot(m);
    int my = pos + __popc(mask & ((1u << lane) - 1u));
    if (m) {
      rowlist[my] = t;
      rowpos[2 * t + s] = my;
    }
    pos += __popc(mask);
  }
}

__global__ void moe_combine_kernel(float* __restrict__ h,
                                   const float* __restrict__ out2,
                                   const int* __restrict__ rowpos,
                                   const float* __restrict__ topg) {
  long long t = blockIdx.x;
  int r0 = rowpos[2 * t], r1 = rowpos[2 * t + 1];
  float g0 = topg[2 * t], g1 = topg[2 * t + 1];
  for (int c = threadIdx.x; c < kD; c += 256)
    h[t * kD + c] += g0 * out2[(long long)r0 * kD + c] + g1 * out2[(long long)r1 * kD + c];
}

__global__ void clsidx_kernel(int* __restrict__ idx) {
  int t = threadIdx.x;
  if (t < kB) idx[t] = t * kSEQ;
}

// ---------------------------------------------------------------------------
// Host side
// ---------------------------------------------------------------------------
static inline GemmParams gp_default() {
  GemmParams p{};            // zero-init
  p.div = 1;
  p.alpha = 1.0f;
  return p;
}

template <int MODE, int TRANSB, int ACT>
static inline void launch_gemm_t(const GemmParams& p, int Mmax, int Z, hipStream_t s) {
  dim3 grid((unsigned)((p.N + 127) / 128), (unsigned)((Mmax + 127) / 128), (unsigned)Z);
  gemm_wmma_kernel<MODE, TRANSB, ACT><<<grid, 256, 0, s>>>(p);
}

static inline void launch_gemm(const GemmParams& p, int Mmax, int Z, hipStream_t s) {
  if (p.mode == 0 && p.transB == 0 && p.act == 0)      launch_gemm_t<0, 0, 0>(p, Mmax, Z, s);
  else if (p.mode == 0 && p.transB == 0 && p.act == 1) launch_gemm_t<0, 0, 1>(p, Mmax, Z, s);
  else if (p.mode == 0 && p.transB == 1)               launch_gemm_t<0, 1, 0>(p, Mmax, Z, s);
  else if (p.mode == 1 && p.act == 1)                  launch_gemm_t<1, 0, 1>(p, Mmax, Z, s);
  else if (p.mode == 1)                                launch_gemm_t<1, 0, 0>(p, Mmax, Z, s);
  else                                                 launch_gemm_t<2, 0, 0>(p, Mmax, Z, s);
}

extern "C" void kernel_launch(void* const* d_in, const int* in_sizes, int n_in,
                              void* d_out, int out_size, void* d_ws, size_t ws_size,
                              hipStream_t stream) {
  (void)in_sizes; (void)n_in; (void)out_size; (void)ws_size;
  const float* x        = (const float*)d_in[0];
  const float* w_patch  = (const float*)d_in[1];
  const float* b_patch  = (const float*)d_in[2];
  const float* cls_tok  = (const float*)d_in[3];
  const float* pos_emb  = (const float*)d_in[4];
  const float* ln1_g    = (const float*)d_in[5];
  const float* ln1_b    = (const float*)d_in[6];
  const float* w_qkv    = (const float*)d_in[7];
  const float* w_proj   = (const float*)d_in[8];
  const float* b_proj   = (const float*)d_in[9];
  const float* ln2_g    = (const float*)d_in[10];
  const float* ln2_b    = (const float*)d_in[11];
  const float* w_fc1    = (const float*)d_in[12];
  const float* b_fc1    = (const float*)d_in[13];
  const float* w_fc2    = (const float*)d_in[14];
  const float* b_fc2    = (const float*)d_in[15];
  const float* w_gate   = (const float*)d_in[16];
  const float* w_e1     = (const float*)d_in[17];
  const float* b_e1     = (const float*)d_in[18];
  const float* w_e2     = (const float*)d_in[19];
  const float* b_e2     = (const float*)d_in[20];
  const float* ln_f_g   = (const float*)d_in[21];
  const float* ln_f_b   = (const float*)d_in[22];
  const float* w_head   = (const float*)d_in[23];
  const float* b_head   = (const float*)d_in[24];

  // ---- workspace carving ----
  char* ws = (char*)d_ws;
  size_t off = 0;
  auto alloc = [&](size_t bytes) -> void* {
    void* p = ws + off;
    off += (bytes + 255) & ~(size_t)255;
    return p;
  };
  float* patches = (float*)alloc((size_t)kB * kNP * kD * 4);          // 1568 x 768
  float* h       = (float*)alloc((size_t)kT * kD * 4);                // residual stream
  float* xln     = (float*)alloc((size_t)kT * kD * 4);                // LN output / tok scratch
  float* qkv     = (float*)alloc((size_t)kT * 3 * kD * 4);            // 1576 x 2304
  float* Smat    = (float*)alloc((size_t)kB * kNH * kSEQ * kSEQ * 4); // 96 x 197 x 197
  float* attn    = (float*)alloc((size_t)kT * kD * 4);
  float* h1      = (float*)alloc((size_t)(2 * kT) * kHID * 4);        // dense & MoE hidden
  float* out2    = (float*)alloc((size_t)(2 * kT) * kD * 4);          // MoE expert outputs
  int*   topi    = (int*)alloc((size_t)kT * 2 * 4);
  float* topg    = (float*)alloc((size_t)kT * 2 * 4);
  int*   rowlist = (int*)alloc((size_t)kT * 2 * 4);
  int*   rowpos  = (int*)alloc((size_t)kT * 2 * 4);
  int*   mcnt    = (int*)alloc(kE * 4);
  int*   moff    = (int*)alloc(kE * 4);
  int*   clsidx  = (int*)alloc(kB * 4);

  // ---- patch embed ----
  patchify_kernel<<<dim3(kNP, kB), 256, 0, stream>>>(x, patches);
  {
    GemmParams p = gp_default();
    p.A = patches; p.lda = kD;
    p.Bm = w_patch; p.ldb = kD;
    p.bias = b_patch;
    p.C = xln; p.ldc = kD;                 // xln used as tok scratch here
    p.M = kB * kNP; p.N = kD; p.K = kD;
    launch_gemm(p, kB * kNP, 1, stream);
  }
  assemble_h_kernel<<<dim3(kSEQ, kB), 256, 0, stream>>>(xln, cls_tok, pos_emb, h);

  // ---- transformer blocks ----
  for (int i = 0; i < 4; ++i) {
    // LN1
    layernorm_kernel<<<kT, 256, 0, stream>>>(h, ln1_g + i * kD, ln1_b + i * kD, xln);
    // QKV (no bias in reference)
    {
      GemmParams p = gp_default();
      p.A = xln; p.lda = kD;
      p.Bm = w_qkv + (long long)i * kD * 3 * kD; p.ldb = 3 * kD;
      p.C = qkv; p.ldc = 3 * kD;
      p.M = kT; p.N = 3 * kD; p.K = kD;
      launch_gemm(p, kT, 1, stream);
    }
    // S = (Q @ K^T) / 8   -- batched over (b, head), B transposed
    {
      GemmParams p = gp_default();
      p.A = qkv; p.lda = 3 * kD; p.ab_b = (long long)kSEQ * 3 * kD; p.ab_s = kHD;
      p.Bm = qkv + kD; p.ldb = 3 * kD; p.bb_b = (long long)kSEQ * 3 * kD; p.bb_s = kHD;
      p.transB = 1;
      p.C = Smat; p.ldc = kSEQ;
      p.cb_b = (long long)kNH * kSEQ * kSEQ; p.cb_s = (long long)kSEQ * kSEQ;
      p.M = kSEQ; p.N = kSEQ; p.K = kHD;
      p.div = kNH; p.alpha = 0.125f;   // 1/sqrt(64)
      launch_gemm(p, kSEQ, kB * kNH, stream);
    }
    softmax_kernel<<<dim3(kSEQ, kB * kNH), 256, 0, stream>>>(Smat, kSEQ);
    // O = P @ V
    {
      GemmParams p = gp_default();
      p.A = Smat; p.lda = kSEQ; p.ab_b = (long long)kNH * kSEQ * kSEQ; p.ab_s = (long long)kSEQ * kSEQ;
      p.Bm = qkv + 2 * kD; p.ldb = 3 * kD; p.bb_b = (long long)kSEQ * 3 * kD; p.bb_s = kHD;
      p.C = attn; p.ldc = kD; p.cb_b = (long long)kSEQ * kD; p.cb_s = kHD;
      p.M = kSEQ; p.N = kHD; p.K = kSEQ;
      p.div = kNH;
      launch_gemm(p, kSEQ, kB * kNH, stream);
    }
    // proj + residual into h (in place)
    {
      GemmParams p = gp_default();
      p.A = attn; p.lda = kD;
      p.Bm = w_proj + (long long)i * kD * kD; p.ldb = kD;
      p.bias = b_proj + (long long)i * kD;
      p.R = h; p.ldr = kD;
      p.C = h; p.ldc = kD;
      p.M = kT; p.N = kD; p.K = kD;
      launch_gemm(p, kT, 1, stream);
    }
    // LN2
    layernorm_kernel<<<kT, 256, 0, stream>>>(h, ln2_g + i * kD, ln2_b + i * kD, xln);

    int j = i / 2;
    if ((i & 1) == 0) {
      // dense MLP
      {
        GemmParams p = gp_default();
        p.A = xln; p.lda = kD;
        p.Bm = w_fc1 + (long long)j * kD * kHID; p.ldb = kHID;
        p.bias = b_fc1 + (long long)j * kHID;
        p.act = 1;
        p.C = h1; p.ldc = kHID;
        p.M = kT; p.N = kHID; p.K = kD;
        launch_gemm(p, kT, 1, stream);
      }
      {
        GemmParams p = gp_default();
        p.A = h1; p.lda = kHID;
        p.Bm = w_fc2 + (long long)j * kHID * kD; p.ldb = kD;
        p.bias = b_fc2 + (long long)j * kD;
        p.R = h; p.ldr = kD;
        p.C = h; p.ldc = kD;
        p.M = kT; p.N = kD; p.K = kHID;
        launch_gemm(p, kT, 1, stream);
      }
    } else {
      // MoE: gate -> route -> grouped expert GEMMs -> combine
      gate_topk_kernel<<<kT, 32, 0, stream>>>(xln, w_gate + (long long)j * kD * kE, topi, topg);
      route_kernel<<<1, 256, 0, stream>>>(topi, rowlist, rowpos, mcnt, moff);
      // expert GEMM1: gather rows, gelu
      {
        GemmParams p = gp_default();
        p.A = xln; p.lda = kD;
        p.Bm = w_e1 + (long long)j * kE * kD * kHID; p.bb_b = (long long)kD * kHID; p.ldb = kHID;
        p.bias = b_e1 + (long long)j * kE * kHID; p.bias_b = kHID;
        p.act = 1;
        p.mode = 1; p.row_idx = rowlist; p.mcnt = mcnt; p.moff = moff;
        p.C = h1; p.ldc = kHID;
        p.M = kT; p.N = kHID; p.K = kD;
        launch_gemm(p, kT, kE, stream);
      }
      // expert GEMM2: compacted rows
      {
        GemmParams p = gp_default();
        p.A = h1; p.lda = kHID;
        p.Bm = w_e2 + (long long)j * kE * kHID * kD; p.bb_b = (long long)kHID * kD; p.ldb = kD;
        p.bias = b_e2 + (long long)j * kE * kD; p.bias_b = kD;
        p.mode = 2; p.mcnt = mcnt; p.moff = moff;
        p.C = out2; p.ldc = kD;
        p.M = kT; p.N = kD; p.K = kHID;
        launch_gemm(p, kT, kE, stream);
      }
      moe_combine_kernel<<<kT, 256, 0, stream>>>(h, out2, rowpos, topg);
    }
  }

  // ---- final LN + head on cls rows ----
  layernorm_kernel<<<kT, 256, 0, stream>>>(h, ln_f_g, ln_f_b, xln);
  clsidx_kernel<<<1, 32, 0, stream>>>(clsidx);
  {
    GemmParams p = gp_default();
    p.A = xln; p.lda = kD;
    p.Bm = w_head; p.ldb = kNCLS;
    p.bias = b_head;
    p.mode = 1; p.row_idx = clsidx;
    p.C = (float*)d_out; p.ldc = kNCLS;
    p.M = kB; p.N = kNCLS; p.K = kD;
    launch_gemm(p, kB, 1, stream);
  }
}
